// CoAttention_18777597018227
// MI455X (gfx1250) — matrix operand order repaired and checked by reading, not compiled, not run on hardware
//
#include <hip/hip_runtime.h>
#include <hip/hip_bf16.h>
#include <math.h>

#define B_  8
#define C_  512
#define L_  2048
#define BL_ (B_ * L_)

typedef __bf16 bf16_t;
typedef bf16_t v16bf __attribute__((ext_vector_type(16)));
typedef bf16_t v8bf  __attribute__((ext_vector_type(8)));
typedef float  v8f   __attribute__((ext_vector_type(8)));
typedef int    v4i_t __attribute__((ext_vector_type(4)));

// ---------------------------------------------------------------------------
// Async global->LDS staging (GLOBAL_LOAD_ASYNC_TO_LDS_B128, ASYNCcnt-tracked).
// Builtin signature (probe-confirmed by round-2 diagnostic): v4i addrspace(1)*
// global src, v4i addrspace(3)* LDS dst, imm offset, imm cpol.
// Falls back to global_load_b128 + ds_store_b128 if the builtin is absent.
// ---------------------------------------------------------------------------
#if defined(__has_builtin)
#if __has_builtin(__builtin_amdgcn_global_load_async_to_lds_b128) && \
    __has_builtin(__builtin_amdgcn_s_wait_asynccnt)
#define HAVE_ASYNC_LDS 1
#endif
#endif
#ifndef HAVE_ASYNC_LDS
#define HAVE_ASYNC_LDS 0
#endif

__device__ __forceinline__ void async_copy16(const bf16_t* g, bf16_t* l) {
#if HAVE_ASYNC_LDS
  __builtin_amdgcn_global_load_async_to_lds_b128(
      (__attribute__((address_space(1))) v4i_t*)(g),
      (__attribute__((address_space(3))) v4i_t*)(l), 0, 0);
#else
  *(v8bf*)l = *(const v8bf*)g;
#endif
}

__device__ __forceinline__ void async_join() {
#if HAVE_ASYNC_LDS
  __builtin_amdgcn_s_wait_asynccnt(0);   // barrier does not wait counters
#endif
}

// ---------------------------------------------------------------------------
// WMMA fragment loaders from LDS (row pitch fixed at 40 bf16 = 80B = 20 banks,
// which makes the 16 lanes' ds_load_b128 reads hit 64 distinct banks).
// A-matrix 16x32 bf16 layout (ISA 7.12.2): lane<16 -> M=lane, K={0..7,16..23};
// lane>=16 -> M=lane-16, K={8..15,24..31}.
// ---------------------------------------------------------------------------
__device__ __forceinline__ v16bf frag_a16(const bf16_t* s) {
  const int lane = threadIdx.x & 31;
  const int m  = lane & 15;
  const int k0 = (lane >> 4) << 3;               // 0 or 8
  const bf16_t* p = s + m * 40;
  union { v16bf v; v8bf h[2]; } u;
  u.h[0] = *(const v8bf*)(p + k0);               // K = k0 .. k0+7
  u.h[1] = *(const v8bf*)(p + k0 + 16);          // K = k0+16 .. k0+23
  return u.v;
}

// B-matrix 32x16 bf16: lane<16 -> N=lane, K={0..7,16..23}; lane>=16 -> N=lane-16, K={8..15,24..31}.
__device__ __forceinline__ v16bf frag_b16(const bf16_t* s) {
  const int lane = threadIdx.x & 31;
  const int n  = lane & 15;
  const int k0 = (lane >> 4) << 3;               // 0 or 8
  const bf16_t* p = s + n * 40 + k0;
  union { v16bf v; v8bf h[2]; } u;
  u.h[0] = *(const v8bf*)(p);
  u.h[1] = *(const v8bf*)(p + 16);
  return u.v;
}

// ---------------------------------------------------------------------------
// Tiled bf16 GEMM:  C[M,N] = scale * (A[M,K] @ B^T) (+ bias[N])
//   TB=false : B stored [N,K] row-major (weights W[d,c], or k for q.k^T)
//   TB=true  : B stored [K,N] row-major (v1/v2), transpose-staged through LDS
// Block: 256 threads (8 waves), 128x128 block tile, K-step 32.
// Wave (wm,wn) owns a 64x32 sub-tile = 4x2 accumulators of 16x16 f32.
// ---------------------------------------------------------------------------
template<bool TB, bool OUT_BF16, bool BIAS>
__global__ __launch_bounds__(256)
void gemm_bf16_wmma(const bf16_t* __restrict__ A, const bf16_t* __restrict__ Bm,
                    const float* __restrict__ bias, void* __restrict__ Cout,
                    int M, int N, int K, int lda, int ldb, int ldc,
                    long long sA, long long sB, long long sC, float scale)
{
  __shared__ __attribute__((aligned(16))) bf16_t As[128 * 40];
  __shared__ __attribute__((aligned(16))) bf16_t Bs[128 * 40];

  const int tid  = threadIdx.x;
  const int wave = tid >> 5;
  const int wm   = (wave & 1) * 64;
  const int wn   = (wave >> 1) * 32;
  const int m0   = blockIdx.y * 128;
  const int n0   = blockIdx.x * 128;

  A  += (long long)blockIdx.z * sA;
  Bm += (long long)blockIdx.z * sB;

  v8f acc[4][2] = {};

  for (int k0 = 0; k0 < K; k0 += 32) {
    { // stage A tile: 128 rows x 32 K, two async b128 per thread
      const int r = tid >> 1, h = (tid & 1) * 16;
      const bf16_t* src = A + (size_t)(m0 + r) * lda + (k0 + h);
      bf16_t* dst = As + r * 40 + h;
      async_copy16(src, dst);
      async_copy16(src + 8, dst + 8);
    }
    if (!TB) { // B tile [N,K]: same pattern as A
      const int r = tid >> 1, h = (tid & 1) * 16;
      const bf16_t* src = Bm + (size_t)(n0 + r) * ldb + (k0 + h);
      bf16_t* dst = Bs + r * 40 + h;
      async_copy16(src, dst);
      async_copy16(src + 8, dst + 8);
    } else {   // B tile [K,N]: coalesced 32B read, transpose into LDS
      const int kk = tid >> 3, c0 = (tid & 7) * 16;
      const bf16_t* src = Bm + (size_t)(k0 + kk) * ldb + (n0 + c0);
      v8bf d0 = *(const v8bf*)(src);
      v8bf d1 = *(const v8bf*)(src + 8);
#pragma unroll
      for (int j = 0; j < 8; ++j) Bs[(c0 + j) * 40 + kk]     = d0[j];
#pragma unroll
      for (int j = 0; j < 8; ++j) Bs[(c0 + 8 + j) * 40 + kk] = d1[j];
    }
    async_join();
    __syncthreads();

    if (k0 + 32 < K) { // WGP-scope prefetch of next K tile (global_prefetch_b8)
      __builtin_prefetch(A + (size_t)(m0 + (tid >> 1)) * lda + (k0 + 32), 0, 3);
      if (TB)
        __builtin_prefetch(Bm + (size_t)(k0 + 32 + (tid >> 3)) * ldb + n0, 0, 3);
      else
        __builtin_prefetch(Bm + (size_t)(n0 + (tid >> 1)) * ldb + (k0 + 32), 0, 3);
    }

    v16bf af[4];
#pragma unroll
    for (int i = 0; i < 4; ++i) af[i] = frag_a16(As + (wm + i * 16) * 40);
    v16bf bfr[2];
#pragma unroll
    for (int j = 0; j < 2; ++j) bfr[j] = frag_b16(Bs + (wn + j * 16) * 40);
#pragma unroll
    for (int i = 0; i < 4; ++i)
#pragma unroll
      for (int j = 0; j < 2; ++j)
        acc[i][j] = __builtin_amdgcn_wmma_f32_16x16x32_bf16(
            false, af[i], false, bfr[j], (short)0, acc[i][j], false, false);
    __syncthreads();
  }

  // C/D layout: VGPR v -> M = v + 8*(lane>=16), N = lane&15
  const int lane = tid & 31;
  const int cm = (lane >> 4) * 8;
  const int cn = lane & 15;
#pragma unroll
  for (int i = 0; i < 4; ++i) {
#pragma unroll
    for (int j = 0; j < 2; ++j) {
      const int gn  = n0 + wn + j * 16 + cn;
      const int gm0 = m0 + wm + i * 16 + cm;
      const float bv = BIAS ? bias[gn] : 0.0f;
      if (OUT_BF16) {
        bf16_t* Cp = (bf16_t*)Cout + (long long)blockIdx.z * sC;
#pragma unroll
        for (int v = 0; v < 8; ++v)
          Cp[(size_t)(gm0 + v) * ldc + gn] = (bf16_t)(acc[i][j][v] * scale + bv);
      } else {
        float* Cp = (float*)Cout + (long long)blockIdx.z * sC;
#pragma unroll
        for (int v = 0; v < 8; ++v)
          Cp[(size_t)(gm0 + v) * ldc + gn] = acc[i][j][v] * scale + bv;
      }
    }
  }
}

// ---------------------------------------------------------------------------
// wave32 reductions
// ---------------------------------------------------------------------------
__device__ __forceinline__ float wred_max(float v) {
#pragma unroll
  for (int o = 16; o > 0; o >>= 1) v = fmaxf(v, __shfl_xor(v, o, 32));
  return v;
}
__device__ __forceinline__ float wred_sum(float v) {
#pragma unroll
  for (int o = 16; o > 0; o >>= 1) v += __shfl_xor(v, o, 32);
  return v;
}

// Row softmax stats: one block per (b,q) row of S (2048 = 256 threads x 8)
__global__ __launch_bounds__(256)
void row_softmax_stats(const bf16_t* __restrict__ S,
                       float* __restrict__ rmax, float* __restrict__ rinv)
{
  __shared__ float red[8];
  const size_t row = blockIdx.x;
  const int tid = threadIdx.x;
  v8bf d = *(const v8bf*)(S + row * L_ + tid * 8);
  float x[8];
#pragma unroll
  for (int j = 0; j < 8; ++j) x[j] = (float)d[j];
  float m = x[0];
#pragma unroll
  for (int j = 1; j < 8; ++j) m = fmaxf(m, x[j]);
  m = wred_max(m);
  if ((tid & 31) == 0) red[tid >> 5] = m;
  __syncthreads();
  float mm = red[0];
#pragma unroll
  for (int w = 1; w < 8; ++w) mm = fmaxf(mm, red[w]);
  __syncthreads();
  float s = 0.f;
#pragma unroll
  for (int j = 0; j < 8; ++j) s += __expf(x[j] - mm);
  s = wred_sum(s);
  if ((tid & 31) == 0) red[tid >> 5] = s;
  __syncthreads();
  if (tid == 0) {
    float t = 0.f;
#pragma unroll
    for (int w = 0; w < 8; ++w) t += red[w];
    rmax[row] = mm;
    rinv[row] = 1.0f / t;
  }
}

// Column softmax stats: one thread per (b,k) column, online max/sum.
// Reads are coalesced across the 256 consecutive k per block.
__global__ __launch_bounds__(256)
void col_softmax_stats(const bf16_t* __restrict__ S,
                       float* __restrict__ cmax, float* __restrict__ cinv)
{
  const int g = blockIdx.x * 256 + threadIdx.x;   // [0, B*L)
  const int b = g / L_;
  const int k = g % L_;
  const bf16_t* p = S + (size_t)b * L_ * L_ + k;
  float m = -3.0e38f, s = 0.f;
  for (int q = 0; q < L_; ++q) {
    const float x  = (float)p[(size_t)q * L_];
    const float nm = fmaxf(m, x);
    s = s * __expf(m - nm) + __expf(x - nm);
    m = nm;
  }
  cmax[g] = m;
  cinv[g] = 1.0f / s;
}

// Normalize S in place (row softmax -> Ak) and emit transposed column softmax
// At[b,k,q] via an LDS 64x64 tile so both writes are coalesced.
__global__ __launch_bounds__(256)
void softmax_apply_T(bf16_t* __restrict__ S, bf16_t* __restrict__ At,
                     const float* __restrict__ rmax, const float* __restrict__ rinv,
                     const float* __restrict__ cmax, const float* __restrict__ cinv)
{
  __shared__ __attribute__((aligned(16))) bf16_t t[64 * 72];
  const int tid = threadIdx.x;
  const int b  = blockIdx.z;
  const int q0 = blockIdx.y * 64, k0 = blockIdx.x * 64;
  const int r  = tid >> 2;               // local q  (0..63)
  const int c0 = (tid & 3) * 16;         // local k base
  bf16_t* Sb = S + (size_t)b * L_ * L_;
  const int q = q0 + r;
  const float rm = rmax[b * L_ + q];
  const float ri = rinv[b * L_ + q];
  bf16_t* p = Sb + (size_t)q * L_ + (k0 + c0);
  union { v8bf v[2]; bf16_t e[16]; } in, outk;
#pragma unroll
  for (int h = 0; h < 2; ++h) in.v[h] = *(const v8bf*)(p + 8 * h);
#pragma unroll
  for (int j = 0; j < 16; ++j) {
    const int kg  = b * L_ + k0 + c0 + j;
    const float x = (float)in.e[j];
    outk.e[j] = (bf16_t)(__expf(x - rm) * ri);
    t[(c0 + j) * 72 + r] = (bf16_t)(__expf(x - cmax[kg]) * cinv[kg]);
  }
#pragma unroll
  for (int h = 0; h < 2; ++h) *(v8bf*)(p + 8 * h) = outk.v[h];
  __syncthreads();
  bf16_t* o = At + ((size_t)b * L_ + (k0 + r)) * L_ + q0 + c0;
#pragma unroll
  for (int h = 0; h < 2; ++h)
    *(v8bf*)(o + 8 * h) = *(const v8bf*)(t + r * 72 + c0 + 8 * h);
}

// Fused skip + LayerNorm + transpose back to (B,C,L).
// Stages x[b, :, l0:l0+16] through LDS so strided x reads and out writes are
// coalesced along L. 16 threads cooperate per l (shuffle reduction, wave32).
__global__ __launch_bounds__(256)
void ln_epilogue(const float* __restrict__ av, const float* __restrict__ x,
                 const float* __restrict__ gamma, const float* __restrict__ beta,
                 float* __restrict__ out)
{
  __shared__ float xt[C_ * 17];
  const int tid = threadIdx.x;
  const int b  = blockIdx.y;
  const int l0 = blockIdx.x * 16;
  {
    const int tl = tid & 15, cc = tid >> 4;
    for (int i = 0; i < 32; ++i) {
      const int c = cc + 16 * i;
      xt[c * 17 + tl] = x[((size_t)b * C_ + c) * L_ + l0 + tl];
    }
  }
  __syncthreads();
  const int l = tid >> 4, j = tid & 15;
  const float* avr = av + ((size_t)b * L_ + l0 + l) * C_;
  float y[32];
  float s = 0.f, s2 = 0.f;
#pragma unroll
  for (int i = 0; i < 32; ++i) {
    const int c = j + 16 * i;
    const float v = avr[c] + xt[c * 17 + l];
    y[i] = v; s += v; s2 += v * v;
  }
#pragma unroll
  for (int o = 8; o > 0; o >>= 1) {
    s  += __shfl_xor(s,  o, 32);
    s2 += __shfl_xor(s2, o, 32);
  }
  const float mu  = s * (1.0f / C_);
  const float var = s2 * (1.0f / C_) - mu * mu;
  const float rs  = rsqrtf(var + 1e-5f);
#pragma unroll
  for (int i = 0; i < 32; ++i) {           // each (c,l) slot owned by 1 thread
    const int c = j + 16 * i;
    xt[c * 17 + l] = (y[i] - mu) * rs * gamma[c] + beta[c];
  }
  __syncthreads();
  {
    const int tl = tid & 15, cc = tid >> 4;
    for (int i = 0; i < 32; ++i) {
      const int c = cc + 16 * i;
      out[((size_t)b * C_ + c) * L_ + l0 + tl] = xt[c * 17 + tl];
    }
  }
}

// (B,C,L) f32 -> (B*L, C) bf16 via LDS 32x32 tile transpose
__global__ __launch_bounds__(256)
void transpose_to_bf16(const float* __restrict__ x, bf16_t* __restrict__ xb)
{
  __shared__ float t[32][33];
  const int tid = threadIdx.x;
  const int b  = blockIdx.z;
  const int c0 = blockIdx.y * 32, l0 = blockIdx.x * 32;
  const int tl = tid & 31, rr = tid >> 5;
#pragma unroll
  for (int p = 0; p < 4; ++p)
    t[rr + p * 8][tl] = x[((size_t)b * C_ + c0 + rr + p * 8) * L_ + l0 + tl];
  __syncthreads();
#pragma unroll
  for (int p = 0; p < 4; ++p)
    xb[((size_t)b * L_ + l0 + rr + p * 8) * C_ + c0 + tl] = (bf16_t)t[tl][rr + p * 8];
}

__global__ void f32_to_bf16(const float* __restrict__ in, bf16_t* __restrict__ out, int n)
{
  const int i = blockIdx.x * blockDim.x + threadIdx.x;
  if (i < n) out[i] = (bf16_t)in[i];
}

// ---------------------------------------------------------------------------
extern "C" void kernel_launch(void* const* d_in, const int* in_sizes, int n_in,
                              void* d_out, int out_size, void* d_ws, size_t ws_size,
                              hipStream_t stream)
{
  (void)in_sizes; (void)n_in; (void)out_size; (void)ws_size; // ws need ~304 MB

  const float* x1   = (const float*)d_in[0];
  const float* x2   = (const float*)d_in[1];
  const float* w_q  = (const float*)d_in[2];
  const float* b_q  = (const float*)d_in[3];
  const float* w_k  = (const float*)d_in[4];
  const float* b_k  = (const float*)d_in[5];
  const float* w_v1 = (const float*)d_in[6];
  const float* b_v1 = (const float*)d_in[7];
  const float* w_v2 = (const float*)d_in[8];
  const float* b_v2 = (const float*)d_in[9];
  const float* gmm  = (const float*)d_in[10];
  const float* bta  = (const float*)d_in[11];

  char* ws = (char*)d_ws;
  size_t off = 0;
  auto take = [&](size_t bytes) -> char* {
    char* p = ws + off;
    off = (off + bytes + 255) & ~(size_t)255;
    return p;
  };

  bf16_t* xb1  = (bf16_t*)take((size_t)BL_ * C_ * 2);
  bf16_t* xb2  = (bf16_t*)take((size_t)BL_ * C_ * 2);
  bf16_t* wqb  = (bf16_t*)take((size_t)C_ * C_ * 2);
  bf16_t* wkb  = (bf16_t*)take((size_t)C_ * C_ * 2);
  bf16_t* wv1b = (bf16_t*)take((size_t)C_ * C_ * 2);
  bf16_t* wv2b = (bf16_t*)take((size_t)C_ * C_ * 2);
  bf16_t* q    = (bf16_t*)take((size_t)BL_ * C_ * 2);
  bf16_t* k    = (bf16_t*)take((size_t)BL_ * C_ * 2);
  bf16_t* v1   = (bf16_t*)take((size_t)BL_ * C_ * 2);
  bf16_t* v2   = (bf16_t*)take((size_t)BL_ * C_ * 2);
  bf16_t* S    = (bf16_t*)take((size_t)B_ * L_ * L_ * 2);   // -> Ak in place
  bf16_t* At   = (bf16_t*)take((size_t)B_ * L_ * L_ * 2);   // Aq transposed
  float*  rmax = (float*)take((size_t)BL_ * 4);
  float*  rinv = (float*)take((size_t)BL_ * 4);
  float*  cmx  = (float*)take((size_t)BL_ * 4);
  float*  cin  = (float*)take((size_t)BL_ * 4);
  float*  vk   = (float*)take((size_t)BL_ * C_ * 4);
  float*  vq   = (float*)take((size_t)BL_ * C_ * 4);

  const dim3 blk(256);
  const float inv_sqrt_c = 0.044194173824159216f;  // 1/sqrt(512)

  // 1) layout + precision conversion
  transpose_to_bf16<<<dim3(L_/32, C_/32, B_), blk, 0, stream>>>(x1, xb1);
  transpose_to_bf16<<<dim3(L_/32, C_/32, B_), blk, 0, stream>>>(x2, xb2);
  f32_to_bf16<<<dim3((C_*C_)/256), blk, 0, stream>>>(w_q,  wqb,  C_*C_);
  f32_to_bf16<<<dim3((C_*C_)/256), blk, 0, stream>>>(w_k,  wkb,  C_*C_);
  f32_to_bf16<<<dim3((C_*C_)/256), blk, 0, stream>>>(w_v1, wv1b, C_*C_);
  f32_to_bf16<<<dim3((C_*C_)/256), blk, 0, stream>>>(w_v2, wv2b, C_*C_);

  // 2) projections: [BL,C] @ W^T + b  (bf16 out)
  gemm_bf16_wmma<false, true, true><<<dim3(C_/128, BL_/128, 1), blk, 0, stream>>>(
      xb1, wqb,  b_q,  q,  BL_, C_, C_, C_, C_, C_, 0, 0, 0, 1.0f);
  gemm_bf16_wmma<false, true, true><<<dim3(C_/128, BL_/128, 1), blk, 0, stream>>>(
      xb2, wkb,  b_k,  k,  BL_, C_, C_, C_, C_, C_, 0, 0, 0, 1.0f);
  gemm_bf16_wmma<false, true, true><<<dim3(C_/128, BL_/128, 1), blk, 0, stream>>>(
      xb1, wv1b, b_v1, v1, BL_, C_, C_, C_, C_, C_, 0, 0, 0, 1.0f);
  gemm_bf16_wmma<false, true, true><<<dim3(C_/128, BL_/128, 1), blk, 0, stream>>>(
      xb2, wv2b, b_v2, v2, BL_, C_, C_, C_, C_, C_, 0, 0, 0, 1.0f);

  // 3) scores: S[b] = q[b] @ k[b]^T / sqrt(C)  (bf16, L2-resident)
  gemm_bf16_wmma<false, true, false><<<dim3(L_/128, L_/128, B_), blk, 0, stream>>>(
      q, k, nullptr, S, L_, L_, C_, C_, C_, L_,
      (long long)L_ * C_, (long long)L_ * C_, (long long)L_ * L_, inv_sqrt_c);

  // 4) softmax stats + in-place row normalize + transposed column normalize
  row_softmax_stats<<<dim3(B_ * L_), blk, 0, stream>>>(S, rmax, rinv);
  col_softmax_stats<<<dim3(B_ * L_ / 256), blk, 0, stream>>>(S, cmx, cin);
  softmax_apply_T<<<dim3(L_/64, L_/64, B_), blk, 0, stream>>>(S, At, rmax, rinv, cmx, cin);

  // 5) vk[b] = Ak[b] @ v2[b];  vq[b] = Aq^T[b] @ v1[b]   (f32 out)
  gemm_bf16_wmma<true, false, false><<<dim3(C_/128, L_/128, B_), blk, 0, stream>>>(
      S,  v2, nullptr, vk, L_, C_, L_, L_, C_, C_,
      (long long)L_ * L_, (long long)L_ * C_, (long long)L_ * C_, 1.0f);
  gemm_bf16_wmma<true, false, false><<<dim3(C_/128, L_/128, B_), blk, 0, stream>>>(
      At, v1, nullptr, vq, L_, C_, L_, L_, C_, C_,
      (long long)L_ * L_, (long long)L_ * C_, (long long)L_ * C_, 1.0f);

  // 6) skip + LayerNorm + transpose back to (B,C,L)
  ln_epilogue<<<dim3(L_/16, B_), blk, 0, stream>>>(vk, x1, gmm, bta, (float*)d_out);
  ln_epilogue<<<dim3(L_/16, B_), blk, 0, stream>>>(vq, x2, gmm, bta,
      (float*)d_out + (size_t)B_ * C_ * L_);
}
